// StructureModule_50568944943181
// MI455X (gfx1250) — matrix operand order repaired and checked by reading, not compiled, not run on hardware
//
#include <hip/hip_runtime.h>
#include <cstdint>
#include <cstddef>

typedef __bf16 bf16_t;
typedef __attribute__((ext_vector_type(16))) __bf16 v16bf;
typedef __attribute__((ext_vector_type(8)))  float  v8f;
typedef unsigned int u32;

#define NRES 512
#define CS   384
#define CZ   128
#define HDS  12
#define PQN  4
#define PROJN 1152   // 192 + 384 + 144 + 432
#define CATN  2112   // 192 + 288 + 96 + 1536

// ---------------------------------------------------------------------------
// wave32 reductions
// ---------------------------------------------------------------------------
__device__ __forceinline__ float wsum(float v) {
#pragma unroll
  for (int o = 16; o > 0; o >>= 1) v += __shfl_xor(v, o, 32);
  return v;
}
__device__ __forceinline__ float wmax(float v) {
#pragma unroll
  for (int o = 16; o > 0; o >>= 1) v = fmaxf(v, __shfl_xor(v, o, 32));
  return v;
}

// ---------------------------------------------------------------------------
// Fast A-fragment loader (no bounds checks; requires lda%8==0, kbase%32==0,
// row valid). A layout: lane holds row m=lane&15; element e ->
// K = (lane>>4)*16 + e. Elements 0..15 are 16 consecutive bf16 at
// K = kbase + hi*16 (32 bytes, 16B aligned).
// ---------------------------------------------------------------------------
__device__ __forceinline__ v16bf load_a_frag(const bf16_t* __restrict__ A,
                                             size_t row, int lda, int kbase) {
  const int hi = (threadIdx.x >> 4) & 1;
  union { v16bf v; uint4 q[2]; } ua;
  const uint4* p = (const uint4*)(A + row * (size_t)lda + kbase + hi * 16);
  ua.q[0] = p[0];   // K = kbase + hi*16 + 0..7
  ua.q[1] = p[1];   // K = kbase + hi*16 + 8..15
  return ua.v;
}

__device__ __forceinline__ v16bf load_b_packed(const bf16_t* __restrict__ Wp,
                                               int tn, int ks, int Ksteps) {
  union { v16bf v; uint4 q[2]; } ub;
  const uint4* bp = (const uint4*)(Wp + (((size_t)(tn * Ksteps + ks) * 32 + threadIdx.x) << 4));
  ub.q[0] = bp[0]; ub.q[1] = bp[1];
  return ub.v;
}

// ---------------------------------------------------------------------------
// Generic GEMM: Y[M,N] = act(A[M,K]_bf16 @ Wpacked + bias). Two N-tiles per
// wave (one A fragment feeds two WMMAs). Requires M%16==0, K%32==0, and the
// packed weight to contain an even number of (zero-padded) N-tiles.
// grid = (M/16, ceil(N/32))
// ---------------------------------------------------------------------------
__global__ __launch_bounds__(32) void gemm_packed(
    const bf16_t* __restrict__ A, int lda,
    const bf16_t* __restrict__ Wp,
    const float* __restrict__ bias,
    float* __restrict__ Yf, bf16_t* __restrict__ Ybf, int ldy,
    int M, int N, int K, int relu) {
  const int tm = blockIdx.x;
  const int tn0 = blockIdx.y * 2, tn1 = tn0 + 1;
  const int lane = threadIdx.x, hi = lane >> 4, lo = lane & 15;
  const int Ksteps = K >> 5;
  const size_t row = (size_t)(tm * 16 + lo);
  v8f acc0 = {}, acc1 = {};
  for (int ks = 0; ks < Ksteps; ++ks) {
    v16bf a  = load_a_frag(A, row, lda, ks << 5);
    v16bf b0 = load_b_packed(Wp, tn0, ks, Ksteps);
    v16bf b1 = load_b_packed(Wp, tn1, ks, Ksteps);
    acc0 = __builtin_amdgcn_wmma_f32_16x16x32_bf16(false, a, false, b0, (short)0, acc0,
                                                   false, false);
    acc1 = __builtin_amdgcn_wmma_f32_16x16x32_bf16(false, a, false, b1, (short)0, acc1,
                                                   false, false);
  }
#pragma unroll
  for (int half = 0; half < 2; ++half) {
    const int n = (half ? tn1 : tn0) * 16 + lo;
    if (n < N) {
      float bv = bias ? bias[n] : 0.f;
#pragma unroll
      for (int r = 0; r < 8; ++r) {
        int m = tm * 16 + r + hi * 8;
        float v = (half ? acc1[r] : acc0[r]) + bv;
        if (relu) v = fmaxf(v, 0.f);
        size_t idx = (size_t)m * ldy + n;
        if (Yf)  Yf[idx]  = v;
        if (Ybf) Ybf[idx] = (bf16_t)v;
      }
    }
  }
}

// Pack fp32 weight W[K,N] (row-major) into B-fragment layout (zero padded).
__global__ __launch_bounds__(32) void pack_w(const float* __restrict__ W,
                                             bf16_t* __restrict__ Wp,
                                             int K, int N, int Ksteps) {
  const int kstep = blockIdx.x, tn = blockIdx.y;
  const int lane = threadIdx.x, hi = lane >> 4, lo = lane & 15;
  const int n = tn * 16 + lo;
  bf16_t* dst = Wp + (((size_t)(tn * Ksteps + kstep) * 32 + lane) << 4);
#pragma unroll
  for (int e = 0; e < 16; ++e) {
    int k = (kstep << 5) + hi * 16 + e;
    float v = (k < K && n < N) ? W[(size_t)k * N + n] : 0.f;
    dst[e] = (bf16_t)v;
  }
}

// ---------------------------------------------------------------------------
// Attention logits: qk via WMMA (K=16 zero-padded to 32) + zb + points + mask
// grid = (jtiles=32, itiles=32, H=12)
// ---------------------------------------------------------------------------
__global__ __launch_bounds__(32) void attn_logits(
    const bf16_t* __restrict__ proj_bf,   // [512 x 1152]: q @0, kv @192
    const float* __restrict__ qp,         // [512][48][3]
    const float* __restrict__ kp,         // [512][48][3]
    const float* __restrict__ zb,         // [512*512][12]
    const float* __restrict__ mask,       // [512]
    const float* __restrict__ head_w,     // [12]
    float* __restrict__ logits) {         // [12][512][512]
  const int tj = blockIdx.x, ti = blockIdx.y, h = blockIdx.z;
  const int lane = threadIdx.x, hi = lane >> 4, lo = lane & 15;
  const int i = ti * 16 + lo, j = tj * 16 + lo;

  // A = q[i, h*16 + c] with K(c) 16..31 zero-padded: clamp address, select val
  union { v16bf v; u32 u[8]; } ua;
  const u32* q32 = (const u32*)(proj_bf + (size_t)i * PROJN + h * 16);
#pragma unroll
  for (int p = 0; p < 8; ++p) {
    int c0 = hi * 16 + (p << 1);
    u32 t = q32[(c0 & 15) >> 1];
    ua.u[p] = (c0 < 16) ? t : 0u;
  }
  // B = k[j, c]; lanes hi==1 carry K 16..31 -> zero
  union { v16bf v; u32 u[8]; } ub;
  const u32* k32 = (const u32*)(proj_bf + (size_t)j * PROJN + 192 + h * 32);
#pragma unroll
  for (int p = 0; p < 8; ++p) {
    u32 t = k32[p];
    ub.u[p] = (hi == 0) ? t : 0u;
  }
  v8f acc = {};
  acc = __builtin_amdgcn_wmma_f32_16x16x32_bf16(false, ua.v, false, ub.v, (short)0, acc,
                                                false, false);
  const float hwv = head_w[h];
  const float sp = (hwv > 20.f) ? hwv : log1pf(expf(hwv));
  const float gamma = sp * 0.13608276348795434f;      // sqrt(1/54)
  const float qk_scale = 0.14433756729740643f;        // sqrt(1/48)
  const float zb_scale = 0.5773502691896258f;         // sqrt(1/3)
  const float mj = mask[j];
  float kpv[12];
#pragma unroll
  for (int t = 0; t < 12; ++t) kpv[t] = kp[(size_t)j * 144 + h * 12 + t];
#pragma unroll
  for (int r = 0; r < 8; ++r) {
    int im = ti * 16 + r + hi * 8;
    float d2 = 0.f;
#pragma unroll
    for (int t = 0; t < 12; ++t) {
      float d = qp[(size_t)im * 144 + h * 12 + t] - kpv[t];
      d2 += d * d;
    }
    float val = acc[r] * qk_scale + zb_scale * zb[((size_t)im * NRES + j) * 12 + h]
              - 0.5f * gamma * d2 + 100000.f * (mask[im] * mj - 1.f);
    logits[((size_t)h * NRES + im) * NRES + j] = val;
  }
}

// softmax over j (512), one wave per (i,h) row; output bf16 probs
__global__ __launch_bounds__(32) void softmax_rows(const float* __restrict__ logits,
                                                   bf16_t* __restrict__ probs) {
  const int i = blockIdx.x, h = blockIdx.y, lane = threadIdx.x;
  const float* row = logits + ((size_t)h * NRES + i) * NRES;
  float vals[16], mx = -1e30f;
#pragma unroll
  for (int t = 0; t < 16; ++t) { vals[t] = row[lane + t * 32]; mx = fmaxf(mx, vals[t]); }
  mx = wmax(mx);
  float s = 0.f;
#pragma unroll
  for (int t = 0; t < 16; ++t) { vals[t] = expf(vals[t] - mx); s += vals[t]; }
  s = wsum(s);
  const float inv = 1.f / s;
  bf16_t* prow = probs + ((size_t)h * NRES + i) * NRES;
#pragma unroll
  for (int t = 0; t < 16; ++t) prow[lane + t * 32] = (bf16_t)(vals[t] * inv);
}

// Y[h][i][n] = sum_j probs[h][i][j] * Bsrc[j][boff_base + h*bhstride + n]
// grid = (Mtiles=32, Ntiles, H=12)
__global__ __launch_bounds__(32) void gemm_attn(
    const bf16_t* __restrict__ P, const bf16_t* __restrict__ Bsrc,
    int ldb, int boff_base, int bhstride, int ncols,
    float* __restrict__ Y, int ldy, int y_head_stride) {
  const int tm = blockIdx.x, tn = blockIdx.y, h = blockIdx.z;
  const int lane = threadIdx.x, hi = lane >> 4, lo = lane & 15;
  const bf16_t* A = P + (size_t)h * NRES * NRES;
  const int boff = boff_base + h * bhstride;
  const int n = tn * 16 + lo;
  const int nc = (n < ncols) ? n : 0;               // clamped gather column
  const size_t row = (size_t)(tm * 16 + lo);
  v8f acc = {};
  for (int ks = 0; ks < NRES / 32; ++ks) {
    v16bf a = load_a_frag(A, row, NRES, ks << 5);
    union { v16bf v; u32 u[8]; } ub;
#pragma unroll
    for (int p = 0; p < 8; ++p) {
      int k0 = (ks << 5) + hi * 16 + 2 * p;
      u32 b0 = *(const unsigned short*)&Bsrc[(size_t)k0 * ldb + boff + nc];
      u32 b1 = *(const unsigned short*)&Bsrc[(size_t)(k0 + 1) * ldb + boff + nc];
      u32 w = b0 | (b1 << 16);
      ub.u[p] = (n < ncols) ? w : 0u;
    }
    acc = __builtin_amdgcn_wmma_f32_16x16x32_bf16(false, a, false, ub.v, (short)0, acc,
                                                  false, false);
  }
  float* Yh = Y + (size_t)h * y_head_stride;
  if (n < ncols) {
#pragma unroll
    for (int r = 0; r < 8; ++r) {
      int m = tm * 16 + r + hi * 8;
      Yh[(size_t)m * ldy + n] = acc[r];
    }
  }
}

// opair[i][h][c] = sum_j probs[h][i][j] * z[i][j][c]; grid=(i=512, ntile=8)
__global__ __launch_bounds__(32) void gemm_opair(const bf16_t* __restrict__ probs,
                                                 const bf16_t* __restrict__ zb16,
                                                 float* __restrict__ opair) {
  const int i = blockIdx.x, tn = blockIdx.y;
  const int lane = threadIdx.x, hi = lane >> 4, lo = lane & 15;
  const int n = tn * 16 + lo;
  const int hrowA = (lo < HDS) ? lo : 0;            // clamped A row (head)
  const u32 amask = (lo < HDS) ? 0xFFFFFFFFu : 0u;  // zero rows 12..15
  v8f acc = {};
  for (int ks = 0; ks < NRES / 32; ++ks) {
    union { v16bf v; u32 u[8]; } ua;
    {
      const uint4* p = (const uint4*)(probs + ((size_t)hrowA * NRES + i) * NRES
                                      + (ks << 5) + hi * 16);
      union { uint4 q; u32 w[4]; } t0, t1;
      t0.q = p[0]; t1.q = p[1];
#pragma unroll
      for (int k = 0; k < 4; ++k) { ua.u[k] = t0.w[k] & amask; ua.u[4 + k] = t1.w[k] & amask; }
    }
    union { v16bf v; u32 u[8]; } ub;
#pragma unroll
    for (int p = 0; p < 8; ++p) {
      int j0 = (ks << 5) + hi * 16 + 2 * p;
      u32 b0 = *(const unsigned short*)&zb16[((size_t)i * NRES + j0) * CZ + n];
      u32 b1 = *(const unsigned short*)&zb16[((size_t)i * NRES + j0 + 1) * CZ + n];
      ub.u[p] = b0 | (b1 << 16);
    }
    acc = __builtin_amdgcn_wmma_f32_16x16x32_bf16(false, ua.v, false, ub.v, (short)0, acc,
                                                  false, false);
  }
#pragma unroll
  for (int r = 0; r < 8; ++r) {
    int hrow = r + hi * 8;
    if (hrow < HDS) opair[((size_t)i * HDS + hrow) * CZ + n] = acc[r];
  }
}

// ---------------------------------------------------------------------------
// LayerNorm (optional residual), templated on C so per-lane values stay in
// registers; float4 loads + b128 stores. One wave per row; C%128==0.
// ---------------------------------------------------------------------------
template <int C>
__global__ __launch_bounds__(32) void layernorm_rows_t(
    const float* __restrict__ X, const float* __restrict__ R,
    const float* __restrict__ g, const float* __restrict__ b,
    float* __restrict__ outf, bf16_t* __restrict__ outbf,
    float* __restrict__ copy1, float* __restrict__ copy2) {
  constexpr int PER = C / 128;                 // float4s per lane
  const int row = blockIdx.x, lane = threadIdx.x;
  const float4* x4 = (const float4*)(X + (size_t)row * C);
  const float4* r4 = R ? (const float4*)(R + (size_t)row * C) : nullptr;
  float4 v[PER];
  float s = 0.f;
#pragma unroll
  for (int t = 0; t < PER; ++t) {
    float4 val = x4[lane + t * 32];
    if (r4) {
      float4 rv = r4[lane + t * 32];
      val.x += rv.x; val.y += rv.y; val.z += rv.z; val.w += rv.w;
    }
    v[t] = val;
    s += val.x + val.y + val.z + val.w;
  }
  const float mu = wsum(s) / (float)C;
  float var = 0.f;
#pragma unroll
  for (int t = 0; t < PER; ++t) {
    v[t].x -= mu; v[t].y -= mu; v[t].z -= mu; v[t].w -= mu;
    var += v[t].x * v[t].x + v[t].y * v[t].y + v[t].z * v[t].z + v[t].w * v[t].w;
  }
  var = wsum(var) / (float)C;
  const float inv = rsqrtf(var + 1e-5f);
#pragma unroll
  for (int t = 0; t < PER; ++t) {
    const int c4 = lane + t * 32;
    float4 g4 = ((const float4*)g)[c4];
    float4 b4 = ((const float4*)b)[c4];
    float4 o;
    o.x = v[t].x * inv * g4.x + b4.x;
    o.y = v[t].y * inv * g4.y + b4.y;
    o.z = v[t].z * inv * g4.z + b4.z;
    o.w = v[t].w * inv * g4.w + b4.w;
    const size_t idx4 = (size_t)row * (C / 4) + c4;
    if (outf)  ((float4*)outf)[idx4] = o;
    if (copy1) ((float4*)copy1)[idx4] = o;
    if (copy2) ((float4*)copy2)[idx4] = o;
    if (outbf) {
      union { bf16_t h[4]; uint2 w; } pk;
      pk.h[0] = (bf16_t)o.x; pk.h[1] = (bf16_t)o.y;
      pk.h[2] = (bf16_t)o.z; pk.h[3] = (bf16_t)o.w;
      ((uint2*)outbf)[idx4] = pk.w;
    }
  }
}

// quats -> rot (row-major 3x3)
__global__ void quat_rot_kernel(const float* __restrict__ quats, float* __restrict__ rot) {
  int i = blockIdx.x * blockDim.x + threadIdx.x;
  if (i >= NRES) return;
  float w = quats[i * 4 + 0], x = quats[i * 4 + 1], y = quats[i * 4 + 2], z = quats[i * 4 + 3];
  float* R = rot + (size_t)i * 9;
  R[0] = w*w + x*x - y*y - z*z; R[1] = 2.f*(x*y - w*z); R[2] = 2.f*(x*z + w*y);
  R[3] = 2.f*(x*y + w*z); R[4] = w*w - x*x + y*y - z*z; R[5] = 2.f*(y*z - w*x);
  R[6] = 2.f*(x*z - w*y); R[7] = 2.f*(y*z + w*x); R[8] = w*w - x*x - y*y + z*z;
}

__global__ void init_state(float* quats, float* trans) {
  int i = blockIdx.x * blockDim.x + threadIdx.x;
  if (i >= NRES) return;
  quats[i * 4 + 0] = 1.f; quats[i * 4 + 1] = 0.f; quats[i * 4 + 2] = 0.f; quats[i * 4 + 3] = 0.f;
  trans[i * 3 + 0] = 0.f; trans[i * 3 + 1] = 0.f; trans[i * 3 + 2] = 0.f;
}

// apply rot/trans to projected points; block 192 threads per residue
__global__ void rotate_points(const float* __restrict__ proj_f,
                              const float* __restrict__ rot, const float* __restrict__ trans,
                              float* __restrict__ qp, float* __restrict__ kp,
                              float* __restrict__ vpf, bf16_t* __restrict__ vpbf) {
  const int i = blockIdx.x, t = threadIdx.x;
  const float* R = rot + (size_t)i * 9;
  const float tx = trans[i * 3 + 0], ty = trans[i * 3 + 1], tz = trans[i * 3 + 2];
  float vx, vy, vz;
  if (t < 48) {                                      // query points: cols 576 + comp*48 + hp
    const float* base = proj_f + (size_t)i * PROJN + 576;
    vx = base[t]; vy = base[48 + t]; vz = base[96 + t];
    float ox = R[0]*vx + R[1]*vy + R[2]*vz + tx;
    float oy = R[3]*vx + R[4]*vy + R[5]*vz + ty;
    float oz = R[6]*vx + R[7]*vy + R[8]*vz + tz;
    qp[(size_t)i * 144 + t * 3 + 0] = ox;
    qp[(size_t)i * 144 + t * 3 + 1] = oy;
    qp[(size_t)i * 144 + t * 3 + 2] = oz;
  } else {                                           // kv points: cols 720 + comp*144 + hp
    const int hp = t - 48;                           // hp = h*12 + p
    const float* base = proj_f + (size_t)i * PROJN + 720;
    vx = base[hp]; vy = base[144 + hp]; vz = base[288 + hp];
    float ox = R[0]*vx + R[1]*vy + R[2]*vz + tx;
    float oy = R[3]*vx + R[4]*vy + R[5]*vz + ty;
    float oz = R[6]*vx + R[7]*vy + R[8]*vz + tz;
    int h = hp / 12, p = hp % 12;
    if (p < PQN) {
      size_t o = (size_t)i * 144 + (h * 4 + p) * 3;
      kp[o] = ox; kp[o + 1] = oy; kp[o + 2] = oz;
    } else {
      size_t o = (size_t)i * 288 + h * 24 + (p - PQN) * 3;   // [512][h*24 + pv*3 + x]
      vpf[o] = ox; vpf[o + 1] = oy; vpf[o + 2] = oz;
      vpbf[o] = (bf16_t)ox; vpbf[o + 1] = (bf16_t)oy; vpbf[o + 2] = (bf16_t)oz;
    }
  }
}

// build bf16 concat [o | opt_flat | opt_norm | opair] per residue; block 384
__global__ void assemble_cat(const float* __restrict__ o_buf,    // [h][512][16]
                             const float* __restrict__ opt_buf,  // [h][512][24]
                             const float* __restrict__ opair,    // [512][12][128]
                             const float* __restrict__ rot, const float* __restrict__ trans,
                             bf16_t* __restrict__ cat) {
  const int i = blockIdx.x, t = threadIdx.x;
  bf16_t* c = cat + (size_t)i * CATN;
  if (t < 192) {
    int h = t >> 4, ci = t & 15;
    c[t] = (bf16_t)o_buf[((size_t)h * NRES + i) * 16 + ci];
  } else if (t < 288) {
    int j = t - 192, h = j >> 3, p = j & 7;
    const float* ob = opt_buf + ((size_t)h * NRES + i) * 24 + p * 3;
    float gx = ob[0] - trans[i * 3 + 0];
    float gy = ob[1] - trans[i * 3 + 1];
    float gz = ob[2] - trans[i * 3 + 2];
    const float* R = rot + (size_t)i * 9;
    float lx = R[0]*gx + R[3]*gy + R[6]*gz;          // R^T
    float ly = R[1]*gx + R[4]*gy + R[7]*gz;
    float lz = R[2]*gx + R[5]*gy + R[8]*gz;
    int hp = h * 8 + p;
    c[192 + hp]       = (bf16_t)lx;
    c[192 + 96 + hp]  = (bf16_t)ly;
    c[192 + 192 + hp] = (bf16_t)lz;
    c[480 + hp]       = (bf16_t)sqrtf(lx*lx + ly*ly + lz*lz + 1e-8f);
  }
  for (int idx = t; idx < 1536; idx += 384)
    c[576 + idx] = (bf16_t)opair[(size_t)i * 1536 + idx];
}

// upd = s @ w_bb + b_bb; quaternion/translation update; frames & pos outputs
__global__ __launch_bounds__(32) void backbone_update(
    const float* __restrict__ s, const float* __restrict__ w_bb,
    const float* __restrict__ b_bb, const float* __restrict__ rot,
    const int* __restrict__ restype, const float* __restrict__ lit_pos,
    float* __restrict__ quats, float* __restrict__ trans,
    float* __restrict__ out_frames, float* __restrict__ out_pos) {
  const int i = blockIdx.x, lane = threadIdx.x;
  float acc[6] = {0.f, 0.f, 0.f, 0.f, 0.f, 0.f};
  for (int k = lane; k < CS; k += 32) {
    float sv = s[(size_t)i * CS + k];
#pragma unroll
    for (int j = 0; j < 6; ++j) acc[j] += sv * w_bb[k * 6 + j];
  }
#pragma unroll
  for (int j = 0; j < 6; ++j) acc[j] = wsum(acc[j]);
  if (lane == 0) {
    float u[6];
#pragma unroll
    for (int j = 0; j < 6; ++j) u[j] = acc[j] + b_bb[j];
    float q0 = quats[i*4+0], q1 = quats[i*4+1], q2 = quats[i*4+2], q3 = quats[i*4+3];
    // quat_mul(q, (1, u0, u1, u2))
    float nw = q0       - q1*u[0] - q2*u[1] - q3*u[2];
    float nx = q0*u[0] + q1       + q2*u[2] - q3*u[1];
    float ny = q0*u[1] - q1*u[2] + q2       + q3*u[0];
    float nz = q0*u[2] + q1*u[1] - q2*u[0] + q3;
    float inv = rsqrtf(nw*nw + nx*nx + ny*ny + nz*nz);
    nw *= inv; nx *= inv; ny *= inv; nz *= inv;
    const float* R = rot + (size_t)i * 9;            // old rotation
    float tx = trans[i*3+0] + R[0]*u[3] + R[1]*u[4] + R[2]*u[5];
    float ty = trans[i*3+1] + R[3]*u[3] + R[4]*u[4] + R[5]*u[5];
    float tz = trans[i*3+2] + R[6]*u[3] + R[7]*u[4] + R[8]*u[5];
    float Rn[9];
    Rn[0] = nw*nw + nx*nx - ny*ny - nz*nz; Rn[1] = 2.f*(nx*ny - nw*nz); Rn[2] = 2.f*(nx*nz + nw*ny);
    Rn[3] = 2.f*(nx*ny + nw*nz); Rn[4] = nw*nw - nx*nx + ny*ny - nz*nz; Rn[5] = 2.f*(ny*nz - nw*nx);
    Rn[6] = 2.f*(nx*nz - nw*ny); Rn[7] = 2.f*(ny*nz + nw*nx); Rn[8] = nw*nw - nx*nx - ny*ny + nz*nz;
    const float* L = lit_pos + (size_t)restype[i] * 9;
#pragma unroll
    for (int a = 0; a < 3; ++a) {
      float px = Rn[0]*L[a*3] + Rn[1]*L[a*3+1] + Rn[2]*L[a*3+2] + 10.f * tx;
      float py = Rn[3]*L[a*3] + Rn[4]*L[a*3+1] + Rn[5]*L[a*3+2] + 10.f * ty;
      float pz = Rn[6]*L[a*3] + Rn[7]*L[a*3+1] + Rn[8]*L[a*3+2] + 10.f * tz;
      out_pos[(size_t)i * 9 + a * 3 + 0] = px;
      out_pos[(size_t)i * 9 + a * 3 + 1] = py;
      out_pos[(size_t)i * 9 + a * 3 + 2] = pz;
    }
    out_frames[i*7+0] = nw; out_frames[i*7+1] = nx; out_frames[i*7+2] = ny; out_frames[i*7+3] = nz;
    out_frames[i*7+4] = 10.f*tx; out_frames[i*7+5] = 10.f*ty; out_frames[i*7+6] = 10.f*tz;
    quats[i*4+0] = nw; quats[i*4+1] = nx; quats[i*4+2] = ny; quats[i*4+3] = nz;
    trans[i*3+0] = tx; trans[i*3+1] = ty; trans[i*3+2] = tz;
  }
}

__global__ void concat_bias(const float* b0, const float* b1, const float* b2,
                            const float* b3, float* out) {
  int t = blockIdx.x * blockDim.x + threadIdx.x;
  if (t < 192) out[t] = b0[t];
  else if (t < 576) out[t] = b1[t - 192];
  else if (t < 720) out[t] = b2[t - 576];
  else if (t < PROJN) out[t] = b3[t - 720];
}

// ---------------------------------------------------------------------------
// host: orchestration
// ---------------------------------------------------------------------------
extern "C" void kernel_launch(void* const* d_in, const int* in_sizes, int n_in,
                              void* d_out, int out_size, void* d_ws, size_t ws_size,
                              hipStream_t stream) {
  const float* in_single = (const float*)d_in[0];
  const float* in_pair   = (const float*)d_in[1];
  const int*   restype   = (const int*)d_in[2];
  const float* mask      = (const float*)d_in[3];
  const float* ln_s_g = (const float*)d_in[4],  *ln_s_b = (const float*)d_in[5];
  const float* ln_z_g = (const float*)d_in[6],  *ln_z_b = (const float*)d_in[7];
  const float* w_in = (const float*)d_in[8],    *b_in = (const float*)d_in[9];
  const float* wq   = (const float*)d_in[10],   *bq   = (const float*)d_in[11];
  const float* wkv  = (const float*)d_in[12],   *bkv  = (const float*)d_in[13];
  const float* wqp  = (const float*)d_in[14],   *bqp  = (const float*)d_in[15];
  const float* wkvp = (const float*)d_in[16],   *bkvp = (const float*)d_in[17];
  const float* wb   = (const float*)d_in[18],   *bb_z = (const float*)d_in[19];
  const float* head_w = (const float*)d_in[20];
  const float* wo   = (const float*)d_in[21],   *bo   = (const float*)d_in[22];
  const float* ln_ipa_g = (const float*)d_in[23], *ln_ipa_b = (const float*)d_in[24];
  const float* t_w1 = (const float*)d_in[25],   *t_b1 = (const float*)d_in[26];
  const float* t_w2 = (const float*)d_in[27],   *t_b2 = (const float*)d_in[28];
  const float* t_w3 = (const float*)d_in[29],   *t_b3 = (const float*)d_in[30];
  const float* ln_t_g = (const float*)d_in[31], *ln_t_b = (const float*)d_in[32];
  const float* w_bb = (const float*)d_in[33],   *b_bb = (const float*)d_in[34];
  const float* lit_pos = (const float*)d_in[35];

  float* out = (float*)d_out;
  float* out_frames = out;                         // (8,1,512,7)
  float* out_pos    = out + 8 * NRES * 7;          // (8,1,512,3,3)
  float* out_states = out + 8 * NRES * 7 + 8 * NRES * 9;  // (8,1,512,384)
  float* out_sfin   = out_states + 8 * NRES * CS;  // (1,512,384)

  char* wsp = (char*)d_ws;
  auto alloc = [&](size_t bytes) -> char* {
    char* p = wsp; wsp += (bytes + 255) & ~(size_t)255; return p;
  };
  bf16_t* z_bf    = (bf16_t*)alloc((size_t)NRES * NRES * CZ * 2);
  float*  zb      = (float*) alloc((size_t)NRES * NRES * 12 * 4);
  float*  s_f     = (float*) alloc((size_t)NRES * CS * 4);
  bf16_t* s_bf    = (bf16_t*)alloc((size_t)NRES * CS * 2);
  bf16_t* sln_bf  = (bf16_t*)alloc((size_t)NRES * CS * 2);
  float*  proj_f  = (float*) alloc((size_t)NRES * PROJN * 4);
  bf16_t* proj_bf = (bf16_t*)alloc((size_t)NRES * PROJN * 2);
  float*  qp      = (float*) alloc((size_t)NRES * 144 * 4);
  float*  kp      = (float*) alloc((size_t)NRES * 144 * 4);
  float*  vpf     = (float*) alloc((size_t)NRES * 288 * 4);
  bf16_t* vpbf    = (bf16_t*)alloc((size_t)NRES * 288 * 2);
  float*  logits  = (float*) alloc((size_t)HDS * NRES * NRES * 4);
  bf16_t* probs   = (bf16_t*)alloc((size_t)HDS * NRES * NRES * 2);
  float*  o_buf   = (float*) alloc((size_t)HDS * NRES * 16 * 4);
  float*  opt_buf = (float*) alloc((size_t)HDS * NRES * 24 * 4);
  float*  opair   = (float*) alloc((size_t)NRES * HDS * CZ * 4);
  bf16_t* cat_bf  = (bf16_t*)alloc((size_t)NRES * CATN * 2);
  float*  tmp_f   = (float*) alloc((size_t)NRES * CS * 4);
  bf16_t* t1_bf   = (bf16_t*)alloc((size_t)NRES * CS * 2);
  bf16_t* t2_bf   = (bf16_t*)alloc((size_t)NRES * CS * 2);
  float*  quats   = (float*) alloc((size_t)NRES * 4 * 4);
  float*  trans   = (float*) alloc((size_t)NRES * 3 * 4);
  float*  rot     = (float*) alloc((size_t)NRES * 9 * 4);
  float*  bias_pr = (float*) alloc(PROJN * 4);
  // packed weights: Ntiles*Ksteps*512 bf16 elems each (even # of N-tiles)
  bf16_t* win_p  = (bf16_t*)alloc((size_t)24 * 12 * 512 * 2);
  bf16_t* proj_p = (bf16_t*)alloc((size_t)72 * 12 * 512 * 2);
  bf16_t* wo_p   = (bf16_t*)alloc((size_t)24 * 66 * 512 * 2);
  bf16_t* tw1_p  = (bf16_t*)alloc((size_t)24 * 12 * 512 * 2);
  bf16_t* tw2_p  = (bf16_t*)alloc((size_t)24 * 12 * 512 * 2);
  bf16_t* tw3_p  = (bf16_t*)alloc((size_t)24 * 12 * 512 * 2);
  bf16_t* wb_p   = (bf16_t*)alloc((size_t)2 * 4 * 512 * 2);   // padded to 2 N-tiles

  // ---- one-time prep ----
  init_state<<<2, 256, 0, stream>>>(quats, trans);
  pack_w<<<dim3(12, 24), 32, 0, stream>>>(w_in, win_p, 384, 384, 12);
  pack_w<<<dim3(12, 12), 32, 0, stream>>>(wq,   proj_p,                       384, 192, 12);
  pack_w<<<dim3(12, 24), 32, 0, stream>>>(wkv,  proj_p + (size_t)12*12*512,  384, 384, 12);
  pack_w<<<dim3(12,  9), 32, 0, stream>>>(wqp,  proj_p + (size_t)36*12*512,  384, 144, 12);
  pack_w<<<dim3(12, 27), 32, 0, stream>>>(wkvp, proj_p + (size_t)45*12*512,  384, 432, 12);
  pack_w<<<dim3(66, 24), 32, 0, stream>>>(wo,   wo_p, 2112, 384, 66);
  pack_w<<<dim3(12, 24), 32, 0, stream>>>(t_w1, tw1_p, 384, 384, 12);
  pack_w<<<dim3(12, 24), 32, 0, stream>>>(t_w2, tw2_p, 384, 384, 12);
  pack_w<<<dim3(12, 24), 32, 0, stream>>>(t_w3, tw3_p, 384, 384, 12);
  pack_w<<<dim3( 4,  2), 32, 0, stream>>>(wb,   wb_p, 128, 12, 4);
  concat_bias<<<5, 256, 0, stream>>>(bq, bkv, bqp, bkvp, bias_pr);

  // s = LN(single) @ w_in + b_in
  layernorm_rows_t<CS><<<NRES, 32, 0, stream>>>(in_single, nullptr, ln_s_g, ln_s_b,
                                                nullptr, sln_bf, nullptr, nullptr);
  gemm_packed<<<dim3(32, 12), 32, 0, stream>>>(sln_bf, CS, win_p, b_in, s_f, s_bf, CS,
                                               NRES, CS, CS, 0);
  // z = LN(pair) (bf16, L2-resident); zb = z @ wb + bb_z
  layernorm_rows_t<CZ><<<NRES * NRES, 32, 0, stream>>>(in_pair, nullptr, ln_z_g, ln_z_b,
                                                       nullptr, z_bf, nullptr, nullptr);
  gemm_packed<<<dim3(NRES * NRES / 16, 1), 32, 0, stream>>>(z_bf, CZ, wb_p, bb_z, zb,
                                                            nullptr, 12, NRES * NRES, 12, CZ, 0);

  // ---- 8 sequential IPA blocks ----
  for (int l = 0; l < 8; ++l) {
    quat_rot_kernel<<<2, 256, 0, stream>>>(quats, rot);
    gemm_packed<<<dim3(32, 36), 32, 0, stream>>>(s_bf, CS, proj_p, bias_pr,
                                                 proj_f, proj_bf, PROJN, NRES, PROJN, CS, 0);
    rotate_points<<<NRES, 192, 0, stream>>>(proj_f, rot, trans, qp, kp, vpf, vpbf);
    attn_logits<<<dim3(32, 32, HDS), 32, 0, stream>>>(proj_bf, qp, kp, zb, mask, head_w, logits);
    softmax_rows<<<dim3(NRES, HDS), 32, 0, stream>>>(logits, probs);
    gemm_attn<<<dim3(32, 1, HDS), 32, 0, stream>>>(probs, proj_bf, PROJN, 208, 32, 16,
                                                   o_buf, 16, NRES * 16);
    gemm_attn<<<dim3(32, 2, HDS), 32, 0, stream>>>(probs, vpbf, 288, 0, 24, 24,
                                                   opt_buf, 24, NRES * 24);
    gemm_opair<<<dim3(NRES, 8), 32, 0, stream>>>(probs, z_bf, opair);
    assemble_cat<<<NRES, 384, 0, stream>>>(o_buf, opt_buf, opair, rot, trans, cat_bf);
    gemm_packed<<<dim3(32, 12), 32, 0, stream>>>(cat_bf, CATN, wo_p, bo, tmp_f, nullptr, CS,
                                                 NRES, CS, CATN, 0);
    // s = LN_ipa(s + ipa_out)
    layernorm_rows_t<CS><<<NRES, 32, 0, stream>>>(s_f, tmp_f, ln_ipa_g, ln_ipa_b,
                                                  s_f, s_bf, nullptr, nullptr);
    // transition
    gemm_packed<<<dim3(32, 12), 32, 0, stream>>>(s_bf, CS, tw1_p, t_b1, nullptr, t1_bf, CS,
                                                 NRES, CS, CS, 1);
    gemm_packed<<<dim3(32, 12), 32, 0, stream>>>(t1_bf, CS, tw2_p, t_b2, nullptr, t2_bf, CS,
                                                 NRES, CS, CS, 1);
    gemm_packed<<<dim3(32, 12), 32, 0, stream>>>(t2_bf, CS, tw3_p, t_b3, tmp_f, nullptr, CS,
                                                 NRES, CS, CS, 0);
    // s = LN_t(s + t3); also emit states[l] (and final s on last block)
    layernorm_rows_t<CS><<<NRES, 32, 0, stream>>>(s_f, tmp_f, ln_t_g, ln_t_b, s_f, s_bf,
                                                  out_states + (size_t)l * NRES * CS,
                                                  (l == 7) ? out_sfin : nullptr);
    // backbone update + frames/pos outputs
    backbone_update<<<NRES, 32, 0, stream>>>(s_f, w_bb, b_bb, rot, restype, lit_pos,
                                             quats, trans,
                                             out_frames + (size_t)l * NRES * 7,
                                             out_pos + (size_t)l * NRES * 9);
  }
}
